// VanillaLSTM_31568009625830
// MI455X (gfx1250) — matrix operand order repaired and checked by reading, not compiled, not run on hardware
//
#include <hip/hip_runtime.h>
#include <hip/hip_bf16.h>

// ---------------- problem constants ----------------
constexpr int Bc  = 64;     // batch
constexpr int Tc  = 512;    // seq len
constexpr int Nc  = 128;    // note length (input / output width)
constexpr int Hc  = 512;    // lstm size
constexpr int G4H = 4 * Hc; // 2048 gate width
constexpr int BT  = Bc * Tc;        // 32768
constexpr int T1  = Tc + 1;         // h buffer has a zero slab at t=0
constexpr int NWG = 16;             // persistent workgroups (each owns 32 h-cols)

typedef __attribute__((ext_vector_type(16))) _Float16 v16h;
typedef __attribute__((ext_vector_type(8)))  _Float16 v8h;
typedef __attribute__((ext_vector_type(8)))  float    v8f;

// ---------------- workspace layout (bytes, 256-aligned) ----------------
constexpr size_t OFF_ACCUM = 0;            // [0]=loss accum (f32), [1]=grid sync (u32)
constexpr size_t OFF_XF16  = 256;
constexpr size_t SZ_XF16   = (size_t)BT * Nc * 2;            // 8 MB
constexpr size_t OFF_XG    = OFF_XF16 + SZ_XF16;
constexpr size_t SZ_XG     = (size_t)BT * G4H * 4;           // 256 MB fp32
constexpr size_t OFF_HALL  = OFF_XG + SZ_XG;
constexpr size_t SZ_HALL   = (size_t)Bc * T1 * Hc * 2;       // 33.6 MB f16
constexpr size_t OFF_WXP   = OFF_HALL + SZ_HALL;
constexpr size_t SZ_WXP    = (size_t)Nc * G4H * 2;
constexpr size_t OFF_WHP   = OFF_WXP + SZ_WXP;
constexpr size_t SZ_WHP    = (size_t)Hc * G4H * 2;
constexpr size_t OFF_WDP   = OFF_WHP + SZ_WHP;

// ---------------- device helpers ----------------
__device__ __forceinline__ v8f zero8() {
    v8f z = {0.f, 0.f, 0.f, 0.f, 0.f, 0.f, 0.f, 0.f};
    return z;
}

__device__ __forceinline__ v8f wmma_f16(v16h a, v16h b, v8f c) {
    return __builtin_amdgcn_wmma_f32_16x16x32_f16(
        /*neg_a=*/false, a, /*neg_b=*/false, b,
        /*c_mod=*/(short)0, c, /*reuse_a=*/false, /*reuse_b=*/false);
}

// 16x32 f16 A fragment from row-major storage.
// lane L: row = L&15; lanes 0-15 hold K=0..7 & 16..23, lanes 16-31 hold K=8..15 & 24..31.
__device__ __forceinline__ v16h load_a_frag(const _Float16* base, int row0,
                                            size_t rowStride, int k0, int lane) {
    int r   = lane & 15;
    int sel = lane >> 4;
    const _Float16* p = base + (size_t)(row0 + r) * rowStride + k0 + sel * 8;
    v8h lo = *(const v8h*)p;
    v8h hi = *(const v8h*)(p + 16);
    return __builtin_shufflevector(lo, hi, 0, 1, 2, 3, 4, 5, 6, 7,
                                          8, 9, 10, 11, 12, 13, 14, 15);
}

// 32x16 f16 B fragment from pre-packed global buffer.
__device__ __forceinline__ v16h load_b_frag(const _Float16* pack, int ntile,
                                            int kt, int KT, int lane) {
    const _Float16* p = pack + (((size_t)ntile * KT + kt) * 32 + lane) * 16;
    return *(const v16h*)p;
}

__device__ __forceinline__ float sigmoidf_(float x) {
    return 1.0f / (1.0f + __expf(-x));
}

// ---------------- kernels ----------------
__global__ void zero_init_kernel(float* accum, unsigned* syncc, _Float16* hall) {
    int idx = blockIdx.x * blockDim.x + threadIdx.x;
    if (idx == 0) { accum[0] = 0.f; syncc[0] = 0u; }
    if (idx < Bc * Hc) {
        int b = idx / Hc, h = idx - b * Hc;
        hall[(size_t)b * T1 * Hc + h] = (_Float16)0.f;   // t = -1 slab
    }
}

__global__ void convert_x_kernel(const float* __restrict__ X, _Float16* __restrict__ Xh) {
    size_t idx = (size_t)blockIdx.x * blockDim.x + threadIdx.x;
    if (idx < (size_t)BT * Nc) Xh[idx] = (_Float16)X[idx];
}

// Pack row-major f32 W[K][Nn] into B-fragment-ordered f16.
__global__ void pack_b_kernel(const float* __restrict__ W, _Float16* __restrict__ dst,
                              int K, int Nn) {
    size_t idx = (size_t)blockIdx.x * blockDim.x + threadIdx.x;
    size_t total = (size_t)K * Nn;
    if (idx >= total) return;
    int KT   = K >> 5;
    int i    = (int)(idx & 15);
    int lane = (int)((idx >> 4) & 31);
    size_t tile = idx >> 9;
    int kt = (int)(tile % KT);
    int nt = (int)(tile / KT);
    int k  = kt * 32 + (i >> 3) * 16 + (lane >> 4) * 8 + (i & 7);
    int n  = nt * 16 + (lane & 15);
    dst[idx] = (_Float16)W[(size_t)k * Nn + n];
}

// xg[BT][4H] = Xh[BT][N] @ WxPack + bias     (64x64 tile per WG, 8 waves)
__global__ void gemm_xg_kernel(const _Float16* __restrict__ Xh,
                               const _Float16* __restrict__ wxp,
                               const float* __restrict__ bias,
                               float* __restrict__ xg) {
    int lane = threadIdx.x & 31, wave = threadIdx.x >> 5;
    int rt = wave & 3, ch = wave >> 2;
    int row0 = blockIdx.y * 64 + rt * 16;
    int col0 = blockIdx.x * 64 + ch * 32;
    constexpr int KT = Nc / 32;   // 4

    v16h a[KT];
#pragma unroll
    for (int kt = 0; kt < KT; ++kt)
        a[kt] = load_a_frag(Xh, row0, Nc, kt * 32, lane);

    v8f acc[2];
#pragma unroll
    for (int ct = 0; ct < 2; ++ct) acc[ct] = zero8();

#pragma unroll
    for (int kt = 0; kt < KT; ++kt) {
#pragma unroll
        for (int ct = 0; ct < 2; ++ct) {
            v16h bfr = load_b_frag(wxp, (col0 >> 4) + ct, kt, KT, lane);
            acc[ct] = wmma_f16(a[kt], bfr, acc[ct]);
        }
    }

#pragma unroll
    for (int ct = 0; ct < 2; ++ct) {
#pragma unroll
        for (int v = 0; v < 8; ++v) {
            int m = row0 + (lane >> 4) * 8 + v;
            int n = col0 + ct * 16 + (lane & 15);
            xg[(size_t)m * G4H + n] = acc[ct][v] + bias[n];
        }
    }
}

// Persistent LSTM recurrence: 16 WGs x 256 threads, one grid barrier per step.
// Each WG owns 32 h-columns; its Wh slice (4 gates x 32 cols x K=512 = 128 KB f16)
// is staged in LDS once and feeds every WMMA B operand for all 512 steps.
// Each thread persistently owns 8 (batch,h) cell positions -> c stays in registers.
__global__ void lstm_persistent_kernel(const float* __restrict__ xg,
                                       const _Float16* __restrict__ whp,
                                       _Float16* __restrict__ hall,
                                       unsigned* __restrict__ syncc) {
    extern __shared__ _Float16 ldsB[];   // 8 local ntiles x 16 ktiles x 512 halves
    int lane = threadIdx.x & 31, wave = threadIdx.x >> 5;
    int rt = wave & 3;                   // batch-row tile (16 of 64 rows)
    int ch = wave >> 2;                  // which 16-col half of the WG's 32 h-cols
    int rowbase = rt * 16;
    int hcol    = blockIdx.x * 32 + ch * 16 + (lane & 15);
    constexpr int KT = Hc / 32;          // 16
    const size_t Astride = (size_t)T1 * Hc;

    // ---- stage Wh slice into LDS (once) ----
    // local tile lt = g*2 + ct  ->  global ntile = g*32 + blockIdx.x*2 + ct
    for (int c = threadIdx.x; c < 8192; c += 256) {           // 8192 x 16B = 128 KB
        int tileIdx = c >> 6;            // 64 chunks per 1 KB tile
        int rem = (c & 63) * 8;          // half offset inside tile
        int lt = tileIdx >> 4, kt = tileIdx & 15;
        int g = lt >> 1, ct = lt & 1;
        size_t gTile = (size_t)((g * 32 + blockIdx.x * 2 + ct) * KT + kt);
        const uint4* s = (const uint4*)(whp + gTile * 512 + rem);
        uint4* d = (uint4*)(ldsB + (size_t)tileIdx * 512 + rem);
        *d = *s;
    }
    __syncthreads();

    float creg[8];
#pragma unroll
    for (int v = 0; v < 8; ++v) creg[v] = 0.f;

#pragma unroll 1
    for (int t = 0; t < Tc; ++t) {
        const _Float16* Abase = hall + (size_t)t * Hc;  // row m -> (m*T1 + t)*Hc
        v16h a[KT];
#pragma unroll
        for (int kt = 0; kt < KT; ++kt)
            a[kt] = load_a_frag(Abase, rowbase, Astride, kt * 32, lane);

        v8f acc[4];
#pragma unroll
        for (int g = 0; g < 4; ++g) acc[g] = zero8();

#pragma unroll
        for (int kt = 0; kt < KT; ++kt) {
#pragma unroll
            for (int g = 0; g < 4; ++g) {
                const _Float16* p = ldsB + (((size_t)(g * 2 + ch) * KT + kt) * 32 + lane) * 16;
                v16h bfr = *(const v16h*)p;
                acc[g] = wmma_f16(a[kt], bfr, acc[g]);
            }
        }

        // elementwise cell update (gate order i, j, f, o; forget_bias = 1)
#pragma unroll
        for (int v = 0; v < 8; ++v) {
            int m = rowbase + (lane >> 4) * 8 + v;      // batch index
            size_t xb = ((size_t)m * Tc + t) * G4H + hcol;
            float gi = acc[0][v] + xg[xb];
            float gj = acc[1][v] + xg[xb + Hc];
            float gf = acc[2][v] + xg[xb + 2 * Hc];
            float go = acc[3][v] + xg[xb + 3 * Hc];
            float cn = sigmoidf_(gf + 1.0f) * creg[v] + sigmoidf_(gi) * tanhf(gj);
            float hn = sigmoidf_(go) * tanhf(cn);
            creg[v] = cn;
            hall[((size_t)m * T1 + t + 1) * Hc + hcol] = (_Float16)hn;
        }

        // ---- grid-wide phase barrier (monotonic counter, NWG arrivals/step) ----
        __threadfence();                 // flush this thread's h stores to device
        __syncthreads();
        if (threadIdx.x == 0) {
            atomicAdd(syncc, 1u);
            unsigned target = (unsigned)NWG * (unsigned)(t + 1);
            while (__hip_atomic_load(syncc, __ATOMIC_ACQUIRE,
                                     __HIP_MEMORY_SCOPE_AGENT) < target)
                __builtin_amdgcn_s_sleep(1);
        }
        __syncthreads();
    }
}

// Fused: logits = sigmoid(hs @ Wd + bd); accum += (Y - logits)^2
__global__ void loss_kernel(const _Float16* __restrict__ hall,
                            const _Float16* __restrict__ wdp,
                            const float* __restrict__ bd,
                            const float* __restrict__ Yv,
                            float* __restrict__ accum) {
    __shared__ float sred[256];
    int lane = threadIdx.x & 31, wave = threadIdx.x >> 5;
    int rt = wave & 3, ch = wave >> 2;
    int r0 = blockIdx.x * 64;          // 64 (b,t) rows per WG; 64 | T so b fixed
    int b  = r0 >> 9;                  // r0 / Tc
    int t0 = r0 & (Tc - 1);
    constexpr int KT = Hc / 32;        // 16

    const _Float16* Abase = hall + ((size_t)b * T1 + 1 + t0) * Hc;

    v16h a[KT];
#pragma unroll
    for (int kt = 0; kt < KT; ++kt)
        a[kt] = load_a_frag(Abase, rt * 16, (size_t)Hc, kt * 32, lane);

    v8f acc[4];
#pragma unroll
    for (int ct = 0; ct < 4; ++ct) acc[ct] = zero8();

#pragma unroll
    for (int kt = 0; kt < KT; ++kt) {
#pragma unroll
        for (int ct = 0; ct < 4; ++ct) {
            int ntile = ch * 4 + ct;   // cols = ch*64 + ct*16
            v16h bfr = load_b_frag(wdp, ntile, kt, KT, lane);
            acc[ct] = wmma_f16(a[kt], bfr, acc[ct]);
        }
    }

    float part = 0.f;
#pragma unroll
    for (int ct = 0; ct < 4; ++ct) {
#pragma unroll
        for (int v = 0; v < 8; ++v) {
            int r = r0 + rt * 16 + (lane >> 4) * 8 + v;
            int n = ch * 64 + ct * 16 + (lane & 15);
            float logit = sigmoidf_(acc[ct][v] + bd[n]);
            float e = Yv[(size_t)r * Nc + n] - logit;
            part += e * e;
        }
    }

    sred[threadIdx.x] = part;
    __syncthreads();
#pragma unroll
    for (int s = 128; s > 0; s >>= 1) {
        if (threadIdx.x < s) sred[threadIdx.x] += sred[threadIdx.x + s];
        __syncthreads();
    }
    if (threadIdx.x == 0) atomicAdd(accum, sred[0]);
}

__global__ void finalize_kernel(const float* __restrict__ accum, float* __restrict__ out) {
    out[0] = 100.0f * accum[0] * (1.0f / (float)((size_t)BT * Nc));
}

// ---------------- host launcher ----------------
extern "C" void kernel_launch(void* const* d_in, const int* in_sizes, int n_in,
                              void* d_out, int out_size, void* d_ws, size_t ws_size,
                              hipStream_t stream) {
    (void)in_sizes; (void)n_in; (void)out_size; (void)ws_size;
    const float* X    = (const float*)d_in[0];
    const float* Y    = (const float*)d_in[1];
    const float* Wx   = (const float*)d_in[2];
    const float* Wh   = (const float*)d_in[3];
    const float* bias = (const float*)d_in[4];
    const float* Wd   = (const float*)d_in[5];
    const float* bd   = (const float*)d_in[6];
    float* out = (float*)d_out;

    char* ws = (char*)d_ws;
    float*     accum = (float*)(ws + OFF_ACCUM);
    unsigned*  syncc = (unsigned*)(ws + OFF_ACCUM + 4);
    _Float16*  Xh    = (_Float16*)(ws + OFF_XF16);
    float*     xg    = (float*)(ws + OFF_XG);
    _Float16*  hall  = (_Float16*)(ws + OFF_HALL);
    _Float16*  wxp   = (_Float16*)(ws + OFF_WXP);
    _Float16*  whp   = (_Float16*)(ws + OFF_WHP);
    _Float16*  wdp   = (_Float16*)(ws + OFF_WDP);

    zero_init_kernel<<<(Bc * Hc + 255) / 256, 256, 0, stream>>>(accum, syncc, hall);
    convert_x_kernel<<<(int)(((size_t)BT * Nc + 255) / 256), 256, 0, stream>>>(X, Xh);
    pack_b_kernel<<<(Nc * G4H + 255) / 256, 256, 0, stream>>>(Wx, wxp, Nc, G4H);
    pack_b_kernel<<<(Hc * G4H + 255) / 256, 256, 0, stream>>>(Wh, whp, Hc, G4H);
    pack_b_kernel<<<(Hc * Nc + 255) / 256, 256, 0, stream>>>(Wd, wdp, Hc, Nc);

    gemm_xg_kernel<<<dim3(G4H / 64, BT / 64), 256, 0, stream>>>(Xh, wxp, bias, xg);

    // persistent recurrence: 16 WGs, 128 KB dynamic LDS each (Wh slice resident)
    lstm_persistent_kernel<<<dim3(NWG), 256, 128 * 1024, stream>>>(xg, whp, hall, syncc);

    loss_kernel<<<dim3(BT / 64), 256, 0, stream>>>(hall, wdp, bd, Y, accum);
    finalize_kernel<<<1, 1, 0, stream>>>(accum, out);
}
